// AllAtomPathAggregator_20134806683945
// MI455X (gfx1250) — matrix operand order repaired and checked by reading, not compiled, not run on hardware
//
#include <hip/hip_runtime.h>

typedef __attribute__((ext_vector_type(16))) _Float16 v16h;
typedef __attribute__((ext_vector_type(8)))  float    v8f;

#define CUT 5.0f
#define RBF_N 32

union F16x16 { v16h v; _Float16 h[16]; };
union F32x8  { v8f  v; float    f[8];  };

__device__ __forceinline__ float silu_f(float x) { return x / (1.0f + __expf(-x)); }
__device__ __forceinline__ float coscut(float r) {
  float w = 0.5f * (__cosf(0.6283185307179586f * r) + 1.0f);
  return (r < CUT) ? w : 0.0f;
}
__device__ __forceinline__ v8f zero8() {
  v8f z = {0.f, 0.f, 0.f, 0.f, 0.f, 0.f, 0.f, 0.f};
  return z;
}
__device__ __forceinline__ v8f wmma_f16(v16h a, v16h b, v8f c) {
  // 8 args: (neg_a, A, neg_b, B, c_mod, C, reuse_a, reuse_b)
  return __builtin_amdgcn_wmma_f32_16x16x32_f16(false, a, false, b, (short)0, c, false, false);
}

// CDNA5 async global->LDS copy (ASYNCcnt-tracked, no VGPR round trip).
// lds_off: wave-relative LDS byte address (= low 32 bits of generic LDS pointer).
__device__ __forceinline__ void async_copy_b128(unsigned lds_off, const void* src) {
  asm volatile("global_load_async_to_lds_b128 %0, %1, off"
               :: "v"(lds_off), "v"(src)
               : "memory");
}
__device__ __forceinline__ void wait_async0() {
  asm volatile("s_wait_asynccnt 0" ::: "memory");
}

// A-fragment (16-bit A 16x32 layout): lane m = lane&15; per K-tile the lane's 16
// halves are two contiguous 8-half chunks at K = kbase+klo+[0..7] and kbase+16+klo+[0..7],
// klo = 8 for lanes 16..31.
__device__ __forceinline__ v16h load_a_lds(const _Float16* s, int ld, int m, int kbase, int klo) {
  F16x16 A;
  const _Float16* row = s + m * ld + kbase;
#pragma unroll
  for (int i = 0; i < 8; i++) A.h[i] = row[klo + i];
#pragma unroll
  for (int i = 0; i < 8; i++) A.h[8 + i] = row[16 + klo + i];
  return A.v;
}

// B fragments pre-swizzled into blocks of 512 halves: element (lane*16 + i) of
// tile t holds W[kt*32 + (lane>>4)*8 + (i<8 ? i : 8+i)][nt*16 + (lane&15)].
__device__ __forceinline__ v16h load_b(const _Float16* frags, int tile, int lane) {
  return *reinterpret_cast<const v16h*>(frags + (size_t)tile * 512 + lane * 16);
}

// ---------------- weight repack: f32 row-major (K,N) -> f16 B-fragment tiles ----
__global__ void k_pack_b(const float* __restrict__ W, _Float16* __restrict__ dst,
                         int K, int N, int ntiles) {
  int tile = blockIdx.x;
  int kt = tile / ntiles, nt = tile % ntiles;
  int lane = threadIdx.x;
  _Float16* out = dst + (size_t)tile * 512 + lane * 16;
  int n = nt * 16 + (lane & 15);
  int kbase = kt * 32 + ((lane >> 4) << 3);
#pragma unroll
  for (int i = 0; i < 16; i++) {
    int k = kbase + (i < 8 ? i : 8 + i);
    float v = (k < K && n < N) ? W[(size_t)k * N + n] : 0.f;
    out[i] = (_Float16)v;
  }
}

// ---------------- layer-1 factorization tables ----------------------------------
__global__ void k_tables(const float* __restrict__ z_emb, const float* __restrict__ e_feat,
                         const float* __restrict__ peW1, const float* __restrict__ peb1,
                         float* __restrict__ A_tab, float* __restrict__ B_tab,
                         float* __restrict__ E1, int nZ, int nE) {
  int row = blockIdx.x;
  int d = threadIdx.x;  // 0..127
  if (row < nZ) {
    float s = 0.f;
    for (int c = 0; c < 32; c++) s += z_emb[row * 32 + c] * peW1[c * 128 + d];
    A_tab[row * 128 + d] = s;
  } else if (row < 2 * nZ) {
    int z = row - nZ;
    float s = 0.f;
    for (int c = 0; c < 32; c++) s += z_emb[z * 32 + c] * peW1[(32 + c) * 128 + d];
    B_tab[z * 128 + d] = s;
  } else {
    int e = row - 2 * nZ;
    if (e < nE) {
      float s = peb1[d];
      for (int c = 0; c < 32; c++) s += e_feat[e * 32 + c] * peW1[(64 + c) * 128 + d];
      E1[e * 128 + d] = s;
    }
  }
}

__global__ void k_pre1(const int* __restrict__ zf, const int* __restrict__ pj,
                       const int* __restrict__ pk, const float* __restrict__ A_tab,
                       const float* __restrict__ B_tab, float* __restrict__ pre1, int P) {
  int idx = blockIdx.x * blockDim.x + threadIdx.x;
  int p = idx >> 7, d = idx & 127;
  if (p < P) pre1[idx] = A_tab[zf[pj[p]] * 128 + d] + B_tab[zf[pk[p]] * 128 + d];
}

__global__ void k_cut(const float* __restrict__ r0j, const float* __restrict__ r0k,
                      const float* __restrict__ rjk, const int* __restrict__ cen,
                      float* __restrict__ cutw, float* __restrict__ norm, int P) {
  int p = blockIdx.x * blockDim.x + threadIdx.x;
  if (p < P) {
    float w = coscut(r0j[p]) * coscut(r0k[p]) * coscut(rjk[p]);
    cutw[p] = w;
    atomicAdd(&norm[cen[p]], w);
  }
}

__global__ void k_zero(float* __restrict__ a, size_t n) {
  size_t i = (size_t)blockIdx.x * blockDim.x + threadIdx.x;
  size_t stride = (size_t)gridDim.x * blockDim.x;
  for (; i < n; i += stride) a[i] = 0.f;
}

// ---------------- geometry MLP: (P,384pad) -> 256 -> 256 -> 32 ------------------
__global__ __launch_bounds__(128) void k_geom(
    const float* __restrict__ h_flat, const int* __restrict__ pj, const int* __restrict__ pk,
    const float* __restrict__ r0j, const float* __restrict__ r0k, const float* __restrict__ rjk,
    const float* __restrict__ cosang, const _Float16* __restrict__ gW1h,
    const _Float16* __restrict__ gW2h, const _Float16* __restrict__ gW3h,
    const float* __restrict__ gb1, const float* __restrict__ gb2, const float* __restrict__ gb3,
    float* __restrict__ ggeom, int P) {
  __shared__ __align__(32) _Float16 sX[16 * 384];
  __shared__ __align__(32) _Float16 sH1[16 * 256];
  __shared__ __align__(32) _Float16 sH2[16 * 256];
  int tid = threadIdx.x;
  int p0 = blockIdx.x * 16;
  const float width = CUT / (RBF_N - 1);
  const float coeff = -0.5f / (width * width);
  for (int idx = tid; idx < 16 * 384; idx += 128) {
    int m = idx / 384, col = idx % 384;
    int p = p0 + m;
    float v = 0.f;
    if (p < P) {
      if (col < 128) v = h_flat[(size_t)pj[p] * 128 + col];
      else if (col < 256) v = h_flat[(size_t)pk[p] * 128 + (col - 128)];
      else if (col < 352) {
        int rb = col - 256, g = rb >> 5, i = rb & 31;
        float r = (g == 0) ? r0j[p] : (g == 1 ? r0k[p] : rjk[p]);
        r = fminf(r, CUT);
        float d = r - (float)i * width;
        v = __expf(coeff * d * d);
      } else if (col == 352) v = cosang[p];
    }
    sX[idx] = (_Float16)v;
  }
  __syncthreads();
  int w = tid >> 5, lane = tid & 31;
  int m_c = ((lane >> 4) << 3);        // C/D layout row offset per VGPR r
  int klo = (lane >= 16) ? 8 : 0;
  int mA = lane & 15;
  {  // layer 1: 384 -> 256, wave w owns columns [w*64, w*64+64)
    v8f C[4] = {zero8(), zero8(), zero8(), zero8()};
    for (int kt = 0; kt < 12; kt++) {
      v16h A = load_a_lds(sX, 384, mA, kt * 32, klo);
#pragma unroll
      for (int j = 0; j < 4; j++)
        C[j] = wmma_f16(A, load_b(gW1h, kt * 16 + (w * 4 + j), lane), C[j]);
    }
#pragma unroll
    for (int j = 0; j < 4; j++) {
      int n = (w * 4 + j) * 16 + (lane & 15);
      float b = gb1[n];
      F32x8 R; R.v = C[j];
#pragma unroll
      for (int r = 0; r < 8; r++) sH1[(r + m_c) * 256 + n] = (_Float16)silu_f(R.f[r] + b);
    }
  }
  __syncthreads();
  {  // layer 2: 256 -> 256
    v8f C[4] = {zero8(), zero8(), zero8(), zero8()};
    for (int kt = 0; kt < 8; kt++) {
      v16h A = load_a_lds(sH1, 256, mA, kt * 32, klo);
#pragma unroll
      for (int j = 0; j < 4; j++)
        C[j] = wmma_f16(A, load_b(gW2h, kt * 16 + (w * 4 + j), lane), C[j]);
    }
#pragma unroll
    for (int j = 0; j < 4; j++) {
      int n = (w * 4 + j) * 16 + (lane & 15);
      float b = gb2[n];
      F32x8 R; R.v = C[j];
#pragma unroll
      for (int r = 0; r < 8; r++) sH2[(r + m_c) * 256 + n] = (_Float16)silu_f(R.f[r] + b);
    }
  }
  __syncthreads();
  if (w < 2) {  // layer 3: 256 -> 32 (linear)
    v8f C = zero8();
    for (int kt = 0; kt < 8; kt++) {
      v16h A = load_a_lds(sH2, 256, mA, kt * 32, klo);
      C = wmma_f16(A, load_b(gW3h, kt * 2 + w, lane), C);
    }
    int n = w * 16 + (lane & 15);
    float b = gb3[n];
    F32x8 R; R.v = C;
#pragma unroll
    for (int r = 0; r < 8; r++) {
      int p = p0 + r + m_c;
      if (p < P) ggeom[(size_t)p * 32 + n] = R.f[r] + b;
    }
  }
}

// ---------------- dominant kernel: element-pair x energy MLP + scatter ----------
// 8 waves / 8 paths per block; dynamic LDS partition (no static LDS -> base = 0):
#define ELEM_PATHS   8
#define ELEM_OFF_W2  0u        // 32 tiles * 1024 B            = 32768
#define ELEM_OFF_W3  32768u    // 8 tiles * 1024 B             =  8192
#define ELEM_OFF_E1  40960u    // nE(80) * 128 * 4 B           = 40960
#define ELEM_OFF_PRE 81920u    // 8 * 128 * 4 B                =  4096
#define ELEM_OFF_GW  86016u    // 8 * 32 * 4 B                 =  1024
#define ELEM_OFF_CEN 87040u    // 8 * 4 B (pad to 128)         =   128
#define ELEM_OFF_H2  87168u    // 8 * 16*128 * 2 B             = 32768
#define ELEM_LDS_SZ  119936u

__global__ __launch_bounds__(256) void k_elem(
    const float* __restrict__ pre1, const float* __restrict__ E1,
    const float* __restrict__ ggeom, const float* __restrict__ cutw,
    const int* __restrict__ cen, const _Float16* __restrict__ peW2h,
    const _Float16* __restrict__ peW3h, const float* __restrict__ pb2,
    const float* __restrict__ pb3, float* __restrict__ agg, int P, int nE) {
  extern __shared__ float smem_f[];
  char* smem = (char*)smem_f;
  _Float16* sW2 = (_Float16*)(smem + ELEM_OFF_W2);
  _Float16* sW3 = (_Float16*)(smem + ELEM_OFF_W3);
  const float* sE1 = (const float*)(smem + ELEM_OFF_E1);
  float* sPre = (float*)(smem + ELEM_OFF_PRE);
  float* sGW = (float*)(smem + ELEM_OFF_GW);
  int* sCen = (int*)(smem + ELEM_OFF_CEN);

  int tid = threadIdx.x;
  int p0 = blockIdx.x * ELEM_PATHS;
  unsigned lds_base = (unsigned)(size_t)smem;  // low 32 bits = LDS byte offset

  // Async DMA staging: weights + E1 go straight to LDS (ASYNCcnt-tracked).
  for (int i = tid; i < 2048; i += 256)  // sW2: 32 KB
    async_copy_b128(lds_base + ELEM_OFF_W2 + i * 16, (const char*)peW2h + i * 16);
  for (int i = tid; i < 512; i += 256)   // sW3: 8 KB
    async_copy_b128(lds_base + ELEM_OFF_W3 + i * 16, (const char*)peW3h + i * 16);
  int e1chunks = (nE * 128 * 4) >> 4;    // 40 KB for nE=80
  for (int i = tid; i < e1chunks; i += 256)
    async_copy_b128(lds_base + ELEM_OFF_E1 + i * 16, (const char*)E1 + i * 16);

  // Per-path staging through VGPRs (needs arithmetic anyway).
  for (int i = tid; i < ELEM_PATHS * 128; i += 256) {
    int p = p0 + (i >> 7);
    sPre[i] = (p < P) ? pre1[(size_t)p * 128 + (i & 127)] : 0.f;
  }
  if (tid < ELEM_PATHS * 32) {
    int p = p0 + (tid >> 5);
    sGW[tid] = (p < P) ? ggeom[(size_t)p * 32 + (tid & 31)] * cutw[p] : 0.f;
  }
  if (tid < ELEM_PATHS) sCen[tid] = (p0 + tid < P) ? cen[p0 + tid] : 0;

  wait_async0();
  __syncthreads();

  int w = tid >> 5, lane = tid & 31;
  int p = p0 + w;
  if (p >= P) return;  // no barriers past this point
  _Float16* myH2 = (_Float16*)(smem + ELEM_OFF_H2) + w * (16 * 128);
  int m_c = ((lane >> 4) << 3);
  int klo = (lane >= 16) ? 8 : 0;
  int e_row = lane & 15;
  int center = sCen[w];
  int etiles = nE >> 4;
  for (int et = 0; et < etiles; et++) {
    const float* e1row = sE1 + (size_t)(et * 16 + e_row) * 128;
    v16h A2[4];
#pragma unroll
    for (int kt = 0; kt < 4; kt++) {  // layer-1: silu(pre1 + E1) -> A fragments
      F16x16 A;
#pragma unroll
      for (int i = 0; i < 8; i++) {
        int k = kt * 32 + klo + i;
        A.h[i] = (_Float16)silu_f(sPre[w * 128 + k] + e1row[k]);
      }
#pragma unroll
      for (int i = 0; i < 8; i++) {
        int k = kt * 32 + 16 + klo + i;
        A.h[8 + i] = (_Float16)silu_f(sPre[w * 128 + k] + e1row[k]);
      }
      A2[kt] = A.v;
    }
#pragma unroll
    for (int nt = 0; nt < 8; nt++) {  // layer-2: 128 -> 128
      v8f C = zero8();
#pragma unroll
      for (int kt = 0; kt < 4; kt++)
        C = wmma_f16(A2[kt], load_b(sW2, kt * 8 + nt, lane), C);
      int n = nt * 16 + (lane & 15);
      float b = pb2[n];
      F32x8 R; R.v = C;
#pragma unroll
      for (int r = 0; r < 8; r++)
        myH2[(r + m_c) * 128 + n] = (_Float16)silu_f(R.f[r] + b);
    }
    v16h A3[4];
#pragma unroll
    for (int kt = 0; kt < 4; kt++) A3[kt] = load_a_lds(myH2, 128, e_row, kt * 32, klo);
#pragma unroll
    for (int nt = 0; nt < 2; nt++) {  // layer-3: 128 -> 32, scale, scatter
      v8f C = zero8();
#pragma unroll
      for (int kt = 0; kt < 4; kt++)
        C = wmma_f16(A3[kt], load_b(sW3, kt * 2 + nt, lane), C);
      int n = nt * 16 + (lane & 15);
      float b = pb3[n];
      float gw = sGW[w * 32 + n];
      F32x8 R; R.v = C;
#pragma unroll
      for (int r = 0; r < 8; r++) {
        int e = et * 16 + r + m_c;
        atomicAdd(&agg[((size_t)center * nE + e) * 32 + n], (R.f[r] + b) * gw);
      }
    }
  }
}

// ---------------- output MLP: (bn*nE, 32) -> 256 -> 128 -------------------------
__global__ __launch_bounds__(128) void k_out(
    const float* __restrict__ agg, const float* __restrict__ norm,
    const _Float16* __restrict__ oW1h, const _Float16* __restrict__ oW2h,
    const float* __restrict__ ob1, const float* __restrict__ ob2,
    float* __restrict__ out, int rows, int nE) {
  __shared__ __align__(32) _Float16 sH[4][16 * 256];
  int tid = threadIdx.x;
  int w = tid >> 5, lane = tid & 31;
  int tile = blockIdx.x * 4 + w;
  int r0 = tile * 16;
  if (r0 >= rows) return;
  int m_c = ((lane >> 4) << 3);
  int klo = (lane >= 16) ? 8 : 0;
  int mA = lane & 15;
  F16x16 A;
  {
    int r = r0 + mA;
    float inv = 1.0f / fmaxf(norm[r / nE], 1e-8f);
    const float* arow = agg + (size_t)r * 32;
#pragma unroll
    for (int i = 0; i < 8; i++) A.h[i] = (_Float16)(arow[klo + i] * inv);
#pragma unroll
    for (int i = 0; i < 8; i++) A.h[8 + i] = (_Float16)(arow[16 + klo + i] * inv);
  }
#pragma unroll
  for (int nt = 0; nt < 16; nt++) {  // layer-1: 32 -> 256
    v8f C = wmma_f16(A.v, load_b(oW1h, nt, lane), zero8());
    int n = nt * 16 + (lane & 15);
    float b = ob1[n];
    F32x8 R; R.v = C;
#pragma unroll
    for (int r = 0; r < 8; r++) sH[w][(r + m_c) * 256 + n] = (_Float16)silu_f(R.f[r] + b);
  }
#pragma unroll
  for (int nt = 0; nt < 8; nt++) {  // layer-2: 256 -> 128
    v8f C = zero8();
#pragma unroll
    for (int kt = 0; kt < 8; kt++)
      C = wmma_f16(load_a_lds(&sH[w][0], 256, mA, kt * 32, klo),
                   load_b(oW2h, kt * 8 + nt, lane), C);
    int n = nt * 16 + (lane & 15);
    float b = ob2[n];
    F32x8 R; R.v = C;
#pragma unroll
    for (int r = 0; r < 8; r++) out[(size_t)(r0 + r + m_c) * 128 + n] = R.f[r] + b;
  }
}

extern "C" void kernel_launch(void* const* d_in, const int* in_sizes, int n_in,
                              void* d_out, int out_size, void* d_ws, size_t ws_size,
                              hipStream_t stream) {
  const float* h_flat = (const float*)d_in[0];
  const int* z_flat = (const int*)d_in[1];
  const float* e_feat = (const float*)d_in[2];
  const int* p_cen = (const int*)d_in[3];
  const int* p_j = (const int*)d_in[4];
  const int* p_k = (const int*)d_in[5];
  const float* p_r0j = (const float*)d_in[6];
  const float* p_r0k = (const float*)d_in[7];
  const float* p_rjk = (const float*)d_in[8];
  const float* p_cos = (const float*)d_in[9];
  // d_in[10]=bsz, d_in[11]=n_atoms: device scalars, dims derived from in_sizes
  const float* z_emb = (const float*)d_in[12];
  const float* peW1 = (const float*)d_in[13];
  const float* peb1 = (const float*)d_in[14];
  const float* peW2 = (const float*)d_in[15];
  const float* peb2 = (const float*)d_in[16];
  const float* peW3 = (const float*)d_in[17];
  const float* peb3 = (const float*)d_in[18];
  const float* gW1 = (const float*)d_in[19];
  const float* gb1 = (const float*)d_in[20];
  const float* gW2 = (const float*)d_in[21];
  const float* gb2 = (const float*)d_in[22];
  const float* gW3 = (const float*)d_in[23];
  const float* gb3 = (const float*)d_in[24];
  const float* oW1 = (const float*)d_in[25];
  const float* ob1 = (const float*)d_in[26];
  const float* oW2 = (const float*)d_in[27];
  const float* ob2 = (const float*)d_in[28];

  int bn = in_sizes[0] / 128;
  int nE = in_sizes[2] / 32;
  int P = in_sizes[3];
  int nZ = in_sizes[12] / 32;

  char* ws = (char*)d_ws;
  size_t off = 0;
  auto alloc = [&](size_t bytes) -> char* {
    char* ptr = ws + off;
    off = (off + bytes + 255) & ~(size_t)255;
    return ptr;
  };
  _Float16* gW1h = (_Float16*)alloc(192 * 512 * 2);
  _Float16* gW2h = (_Float16*)alloc(128 * 512 * 2);
  _Float16* gW3h = (_Float16*)alloc(16 * 512 * 2);
  _Float16* peW2h = (_Float16*)alloc(32 * 512 * 2);
  _Float16* peW3h = (_Float16*)alloc(8 * 512 * 2);
  _Float16* oW1h = (_Float16*)alloc(16 * 512 * 2);
  _Float16* oW2h = (_Float16*)alloc(64 * 512 * 2);
  float* A_tab = (float*)alloc((size_t)nZ * 128 * 4);
  float* B_tab = (float*)alloc((size_t)nZ * 128 * 4);
  float* E1 = (float*)alloc((size_t)nE * 128 * 4);
  float* pre1 = (float*)alloc((size_t)P * 128 * 4);
  float* cutw = (float*)alloc((size_t)P * 4);
  float* ggeom = (float*)alloc((size_t)P * 32 * 4);
  float* agg = (float*)alloc((size_t)bn * nE * 32 * 4);
  float* norm = (float*)alloc((size_t)bn * 4);
  (void)ws_size; (void)n_in; (void)out_size;

  k_zero<<<1024, 256, 0, stream>>>(agg, (size_t)bn * nE * 32);
  k_zero<<<8, 256, 0, stream>>>(norm, (size_t)bn);

  k_pack_b<<<192, 32, 0, stream>>>(gW1, gW1h, 353, 256, 16);
  k_pack_b<<<128, 32, 0, stream>>>(gW2, gW2h, 256, 256, 16);
  k_pack_b<<<16, 32, 0, stream>>>(gW3, gW3h, 256, 32, 2);
  k_pack_b<<<32, 32, 0, stream>>>(peW2, peW2h, 128, 128, 8);
  k_pack_b<<<8, 32, 0, stream>>>(peW3, peW3h, 128, 32, 2);
  k_pack_b<<<16, 32, 0, stream>>>(oW1, oW1h, 32, 256, 16);
  k_pack_b<<<64, 32, 0, stream>>>(oW2, oW2h, 256, 128, 8);

  k_tables<<<2 * nZ + nE, 128, 0, stream>>>(z_emb, e_feat, peW1, peb1, A_tab, B_tab, E1, nZ, nE);
  k_pre1<<<(P * 128 + 255) / 256, 256, 0, stream>>>(z_flat, p_j, p_k, A_tab, B_tab, pre1, P);
  k_cut<<<(P + 255) / 256, 256, 0, stream>>>(p_r0j, p_r0k, p_rjk, p_cen, cutw, norm, P);

  k_geom<<<(P + 15) / 16, 128, 0, stream>>>(h_flat, p_j, p_k, p_r0j, p_r0k, p_rjk, p_cos,
                                            gW1h, gW2h, gW3h, gb1, gb2, gb3, ggeom, P);
  k_elem<<<(P + ELEM_PATHS - 1) / ELEM_PATHS, 256, ELEM_LDS_SZ, stream>>>(
      pre1, E1, ggeom, cutw, p_cen, peW2h, peW3h, peb2, peb3, agg, P, nE);
  int rows = bn * nE;
  k_out<<<(rows + 63) / 64, 128, 0, stream>>>(agg, norm, oW1h, oW2h, ob1, ob2,
                                              (float*)d_out, rows, nE);
}
